// STFormerBlock_23691039605014
// MI455X (gfx1250) — matrix operand order repaired and checked by reading, not compiled, not run on hardware
//
#include <hip/hip_runtime.h>
#include <hip/hip_bf16.h>
#include <math.h>

typedef __attribute__((ext_vector_type(16))) _Float16 v16h;
typedef __attribute__((ext_vector_type(8)))  _Float16 v8h;
typedef __attribute__((ext_vector_type(8)))  float    v8f;

// ---------------- constants ----------------
#define BB   2
#define DD   8
#define HH   112
#define WW   112
#define CC   64
#define C2   32
#define NHD  4
#define NTOK 200704ull            // B*D*H*W
#define NWIN 2048                 // B * (D/2) * (H/7) * (W/7)
#define NSEQ 25088                // B*H*W
#define SCL  0.25f                // (64/4)^-0.5

// ---------------- helpers ----------------
static __device__ __forceinline__ v8f wmma_f16(v16h a, v16h b, v8f c) {
    return __builtin_amdgcn_wmma_f32_16x16x32_f16(false, a, false, b, (short)0, c, false, false);
}

// A-operand (16-bit 16x32) load: per-lane two contiguous 8-half chunks.
static __device__ __forceinline__ v16h load_a(const _Float16* p, int hh) {
    v8h lo = *(const v8h*)(p + 16 * hh);
    v8h hi = *(const v8h*)(p + 16 * hh + 8);
    v16h a;
#pragma unroll
    for (int i = 0; i < 8; ++i) { a[i] = lo[i]; a[i + 8] = hi[i]; }
    return a;
}
static __device__ __forceinline__ v16h load_a_lds(const _Float16* p, int hh) {
    v16h a;
#pragma unroll
    for (int i = 0; i < 8; ++i) { a[i] = p[16 * hh + i]; a[i + 8] = p[16 * hh + 8 + i]; }
    return a;
}

// wave32 XOR-shuffle via ds_swizzle (immediate lane pattern, no address setup)
template <int M>
static __device__ __forceinline__ float swz_xor(float v) {
    return __int_as_float(__builtin_amdgcn_ds_swizzle(__float_as_int(v), (M << 10) | 0x1f));
}
static __device__ __forceinline__ float wave_sum(float v) {
    v += swz_xor<16>(v); v += swz_xor<8>(v); v += swz_xor<4>(v);
    v += swz_xor<2>(v);  v += swz_xor<1>(v);
    return v;
}
static __device__ __forceinline__ float wave_max(float v) {
    v = fmaxf(v, swz_xor<16>(v)); v = fmaxf(v, swz_xor<8>(v)); v = fmaxf(v, swz_xor<4>(v));
    v = fmaxf(v, swz_xor<2>(v));  v = fmaxf(v, swz_xor<1>(v));
    return v;
}

// ---------------- 1. LayerNorm 1 ----------------
__global__ __launch_bounds__(256) void ln1_kernel(const float* __restrict__ x,
                                                  const float* __restrict__ g,
                                                  const float* __restrict__ b,
                                                  _Float16* __restrict__ h16) {
    int wave = threadIdx.x >> 5, lane = threadIdx.x & 31;
    size_t tok = (size_t)blockIdx.x * 8 + wave;
    const float* xp = x + tok * CC;
    float a0 = xp[lane], a1 = xp[lane + 32];
    float m = wave_sum(a0 + a1) * (1.0f / 64.0f);
    float d0 = a0 - m, d1 = a1 - m;
    float var = wave_sum(d0 * d0 + d1 * d1) * (1.0f / 64.0f);
    float r = rsqrtf(var + 1e-5f);
    h16[tok * CC + lane]      = (_Float16)(d0 * r * g[lane] + b[lane]);
    h16[tok * CC + lane + 32] = (_Float16)(d1 * r * g[lane + 32] + b[lane + 32]);
}

// ---------------- 2. generic WMMA GEMM: A[M,KC] x W[KC,NC] + bias ----------------
template <int KC, int NC, bool OUT_F32>
__global__ __launch_bounds__(256) void gemm_kernel(const _Float16* __restrict__ A,
                                                   const float* __restrict__ Wg,
                                                   const float* __restrict__ bias,
                                                   void* __restrict__ Out) {
    __shared__ _Float16 wl[NC * KC];  // column-major [n][k] so B-operand is K-contiguous
    for (int t = threadIdx.x; t < NC * KC; t += 256) {
        int k = t / NC, n = t % NC;
        wl[n * KC + k] = (_Float16)Wg[t];
    }
    __syncthreads();
    int wave = threadIdx.x >> 5, lane = threadIdx.x & 31;
    int hh = lane >> 4, lr = lane & 15;
    size_t mbase = (size_t)blockIdx.x * 128 + wave * 16;
    const _Float16* arow = A + (mbase + lr) * KC;
    v16h av[KC / 32];
#pragma unroll
    for (int kc = 0; kc < KC / 32; ++kc) av[kc] = load_a(arow + kc * 32, hh);
#pragma unroll
    for (int nt = 0; nt < NC / 16; ++nt) {
        v8f acc = {};
#pragma unroll
        for (int kc = 0; kc < KC / 32; ++kc) {
            v16h bm = *(const v16h*)&wl[(nt * 16 + lr) * KC + kc * 32 + hh * 16];
            acc = wmma_f16(av[kc], bm, acc);
        }
        float bv = bias[nt * 16 + lr];
        if (OUT_F32) {
            float* o = (float*)Out;
#pragma unroll
            for (int i = 0; i < 8; ++i)
                o[(mbase + i + 8 * hh) * NC + nt * 16 + lr] = acc[i] + bv;
        } else {
            _Float16* o = (_Float16*)Out;
#pragma unroll
            for (int i = 0; i < 8; ++i)
                o[(mbase + i + 8 * hh) * NC + nt * 16 + lr] = (_Float16)(acc[i] + bv);
        }
    }
}

// ---------------- 3. spatial window attention (one block per window*head) ----------------
__global__ __launch_bounds__(256) void sattn_kernel(const _Float16* __restrict__ qkvs,
                                                    const float* __restrict__ rpb,
                                                    _Float16* __restrict__ attnout) {
    __shared__ _Float16 qs[112 * 32];   // rows padded 98->112, head-dim 16->32 (zeros)
    __shared__ _Float16 ks[112 * 32];
    __shared__ _Float16 vsm[16 * 128];  // V column-major [n][k], K padded 112->128
    __shared__ float    ss[112 * 112];
    __shared__ _Float16 ps[112 * 128];  // first holds bias table, then softmax probs
    __shared__ int      toks[98];

    int head = blockIdx.x & 3;
    int win  = blockIdx.x >> 2;
    int wwi = win & 15, hwi = (win >> 4) & 15, dwi = (win >> 8) & 3, bi = win >> 10;

    // window -> global token table
    for (int t = threadIdx.x; t < 98; t += 256) {
        int id = t / 49, rem = t % 49, ih = rem / 7, iw = rem % 7;
        int dg = dwi * 2 + id, hg = hwi * 7 + ih, wg = wwi * 7 + iw;
        toks[t] = ((bi * DD + dg) * HH + hg) * WW + wg;
    }
    __syncthreads();

    // load q/k/v into LDS with 16-byte vector loads
    for (int t = threadIdx.x; t < 112 * 2; t += 256) {
        int row = t >> 1, c8 = (t & 1) * 8;
        v8h qv8, kv8, vv8, z;
#pragma unroll
        for (int i = 0; i < 8; ++i) z[i] = (_Float16)0.f;
        if (row < 98) {
            const _Float16* base = qkvs + (size_t)toks[row] * 192 + head * 16 + c8;
            qv8 = *(const v8h*)(base);
            kv8 = *(const v8h*)(base + 64);
            vv8 = *(const v8h*)(base + 128);
#pragma unroll
            for (int i = 0; i < 8; ++i) qv8[i] = qv8[i] * (_Float16)0.25f;  // exact pow2 scale
        } else {
            qv8 = z; kv8 = z; vv8 = z;
        }
        *(v8h*)&qs[row * 32 + c8] = qv8;
        *(v8h*)&qs[row * 32 + 16 + c8] = z;
        *(v8h*)&ks[row * 32 + c8] = kv8;
        *(v8h*)&ks[row * 32 + 16 + c8] = z;
#pragma unroll
        for (int i = 0; i < 8; ++i) vsm[(c8 + i) * 128 + row] = vv8[i];
    }
    for (int t = threadIdx.x; t < 16 * 16; t += 256)
        vsm[(t >> 4) * 128 + 112 + (t & 15)] = (_Float16)0.f;

    // relative-position bias table into ps
    for (int t = threadIdx.x; t < 98 * 98; t += 256) {
        int m = t / 98, n = t - m * 98;
        int zm = m / 49, rm = m % 49, ym = rm / 7, xm = rm % 7;
        int zn = n / 49, rn = n % 49, yn = rn / 7, xn = rn % 7;
        int idx = (zm - zn + 1) * 169 + (ym - yn + 6) * 13 + (xm - xn + 6);
        ps[m * 128 + n] = (_Float16)rpb[idx * NHD + head];
    }
    __syncthreads();

    int wave = threadIdx.x >> 5, lane = threadIdx.x & 31;
    int hh = lane >> 4, lr = lane & 15;

    // S = q @ k^T : 7x7 tiles of 16x16, raw accumulators straight to LDS
    for (int tile = wave; tile < 49; tile += 8) {
        int mt = tile / 7, nt = tile % 7;
        v16h a  = load_a_lds(&qs[(mt * 16 + lr) * 32], hh);
        v16h bm = *(const v16h*)&ks[(nt * 16 + lr) * 32 + hh * 16];
        v8f acc = {};
        acc = wmma_f16(a, bm, acc);
        int n = nt * 16 + lr;
#pragma unroll
        for (int i = 0; i < 8; ++i)
            ss[(mt * 16 + i + 8 * hh) * 112 + n] = acc[i];
    }
    __syncthreads();

    // softmax rows with fused bias add + padding mask -> P (f16, padded to 128 cols)
    for (int r = wave; r < 112; r += 8) {
        if (r < 98) {
            float c0 = ss[r * 112 + lane]      + (float)ps[r * 128 + lane];
            float c1 = ss[r * 112 + lane + 32] + (float)ps[r * 128 + lane + 32];
            float c2 = ss[r * 112 + lane + 64] + (float)ps[r * 128 + lane + 64];
            float c3 = (lane < 2) ? (ss[r * 112 + lane + 96] + (float)ps[r * 128 + lane + 96])
                                  : -3e38f;
            float mx = wave_max(fmaxf(fmaxf(c0, c1), fmaxf(c2, c3)));
            float e0 = __expf(c0 - mx), e1 = __expf(c1 - mx), e2 = __expf(c2 - mx);
            float e3 = (lane < 2) ? __expf(c3 - mx) : 0.f;
            float inv = 1.f / wave_sum(e0 + e1 + e2 + e3);
            ps[r * 128 + lane]      = (_Float16)(e0 * inv);
            ps[r * 128 + lane + 32] = (_Float16)(e1 * inv);
            ps[r * 128 + lane + 64] = (_Float16)(e2 * inv);
            ps[r * 128 + lane + 96] = (lane < 2) ? (_Float16)(e3 * inv) : (_Float16)0.f;
        } else {
            ps[r * 128 + lane]      = (_Float16)0.f;
            ps[r * 128 + lane + 32] = (_Float16)0.f;
            ps[r * 128 + lane + 64] = (_Float16)0.f;
            ps[r * 128 + lane + 96] = (_Float16)0.f;
        }
    }
    __syncthreads();

    // out = P @ V, 7 M-tiles, K = 128 (4 chunks)
    if (wave < 7) {
        int mt = wave;
        v8f acc = {};
#pragma unroll
        for (int kc = 0; kc < 4; ++kc) {
            v16h a  = load_a_lds(&ps[(mt * 16 + lr) * 128 + kc * 32], hh);
            v16h bm = *(const v16h*)&vsm[lr * 128 + kc * 32 + hh * 16];
            acc = wmma_f16(a, bm, acc);
        }
#pragma unroll
        for (int i = 0; i < 8; ++i) {
            int m = mt * 16 + i + 8 * hh;
            if (m < 98)
                attnout[(size_t)toks[m] * CC + head * 16 + lr] = (_Float16)acc[i];
        }
    }
}

// ---------------- 4. temporal attention (D=8, head dim 8; VALU) ----------------
__global__ __launch_bounds__(256) void tattn_kernel(const _Float16* __restrict__ qkvt,
                                                    const float* __restrict__ rpbt,
                                                    _Float16* __restrict__ ot) {
    int tid = blockIdx.x * 256 + threadIdx.x;   // NSEQ * NHD * 8 threads
    int dq = tid & 7, head = (tid >> 3) & 3;
    int seq = tid >> 5;
    if (seq >= NSEQ) return;
    int wg = seq % WW, hg = (seq / WW) % HH, bi = seq / (HH * WW);
    auto tokD = [&](int d) -> size_t {
        return (((size_t)(bi * DD + d) * HH + hg) * WW + wg);
    };
    float q[8];
    {
        const _Float16* qp = qkvt + tokD(dq) * 96 + head * 8;
#pragma unroll
        for (int e = 0; e < 8; ++e) q[e] = (float)qp[e] * SCL;
    }
    float s[8], mx = -3e38f;
#pragma unroll
    for (int dk = 0; dk < 8; ++dk) {
        const _Float16* kp = qkvt + tokD(dk) * 96 + 32 + head * 8;
        float acc = 0.f;
#pragma unroll
        for (int e = 0; e < 8; ++e) acc += q[e] * (float)kp[e];
        acc += rpbt[(dq - dk + 7) * NHD + head];
        s[dk] = acc; mx = fmaxf(mx, acc);
    }
    float sum = 0.f;
#pragma unroll
    for (int dk = 0; dk < 8; ++dk) { s[dk] = __expf(s[dk] - mx); sum += s[dk]; }
    float inv = 1.f / sum;
    float o[8] = {0, 0, 0, 0, 0, 0, 0, 0};
#pragma unroll
    for (int dk = 0; dk < 8; ++dk) {
        const _Float16* vp = qkvt + tokD(dk) * 96 + 64 + head * 8;
        float p = s[dk] * inv;
#pragma unroll
        for (int e = 0; e < 8; ++e) o[e] += p * (float)vp[e];
    }
    _Float16* op = ot + tokD(dq) * 32 + head * 8;
#pragma unroll
    for (int e = 0; e < 8; ++e) op[e] = (_Float16)o[e];
}

// ---------------- 5. combine residuals + LayerNorm 2 ----------------
__global__ __launch_bounds__(256) void combine_kernel(const float* __restrict__ x,
                                                      const float* __restrict__ space,
                                                      const float* __restrict__ timeb,
                                                      const float* __restrict__ g,
                                                      const float* __restrict__ b,
                                                      float* __restrict__ xnew,
                                                      _Float16* __restrict__ a1,
                                                      _Float16* __restrict__ a2) {
    int wave = threadIdx.x >> 5, lane = threadIdx.x & 31;
    size_t tok = (size_t)blockIdx.x * 8 + wave;
    size_t base = tok * CC;
    float v0 = x[base + lane] + space[base + lane] + timeb[base + lane];
    float v1 = x[base + lane + 32] + space[base + lane + 32] + timeb[base + lane + 32];
    xnew[base + lane] = v0; xnew[base + lane + 32] = v1;
    float m = wave_sum(v0 + v1) * (1.0f / 64.0f);
    float d0 = v0 - m, d1 = v1 - m;
    float var = wave_sum(d0 * d0 + d1 * d1) * (1.0f / 64.0f);
    float r = rsqrtf(var + 1e-5f);
    a1[tok * C2 + lane] = (_Float16)(d0 * r * g[lane] + b[lane]);
    a2[tok * C2 + lane] = (_Float16)(d1 * r * g[lane + 32] + b[lane + 32]);
}

// ---------------- 6. implicit-GEMM WMMA conv3d 3x3x3, 32->32, pad 1 ----------------
__global__ __launch_bounds__(256) void conv3d_kernel(const _Float16* __restrict__ inA,
                                                     const float* __restrict__ Wg,    // [32][32][27] OIDHW
                                                     const float* __restrict__ bias,
                                                     const _Float16* __restrict__ res,
                                                     _Float16* __restrict__ out,
                                                     _Float16* __restrict__ out2,
                                                     const _Float16* __restrict__ res2,
                                                     int do_lrelu) {
    __shared__ _Float16 wl[27 * 2 * 2 * 256];  // [tap][nt][hh][n][e]
    for (int t = threadIdx.x; t < 27 * 32 * 32; t += 256) {
        int o = t / 864, rem = t % 864, i = rem / 27, tap = rem % 27;
        int nt = o >> 4, n = o & 15, hh2 = i >> 4, e = i & 15;
        wl[(((tap * 2 + nt) * 2 + hh2) << 8) + n * 16 + e] = (_Float16)Wg[t];
    }
    __syncthreads();
    int wave = threadIdx.x >> 5, lane = threadIdx.x & 31;
    int hh = lane >> 4, lr = lane & 15;
    int mt = blockIdx.x * 8 + wave;           // 12544 M-tiles total
    int wt = mt % 7, hc = (mt / 7) % HH, dc = (mt / 784) % DD, bi = mt / 6272;
    int w0 = wt * 16;

    v8f acc0 = {}, acc1 = {};
    for (int tap = 0; tap < 27; ++tap) {
        int kd = tap / 9, kh = (tap / 3) % 3, kw = tap % 3;
        int dd = dc + kd - 1, hr = hc + kh - 1, wv = w0 + lr + kw - 1;
        v16h a;
        if ((unsigned)dd < DD && (unsigned)hr < HH && (unsigned)wv < WW) {
            const _Float16* p = inA + (((size_t)(bi * DD + dd) * HH + hr) * WW + wv) * C2;
            a = load_a(p, hh);
        } else {
#pragma unroll
            for (int i = 0; i < 16; ++i) a[i] = (_Float16)0.f;
        }
        v16h b0 = *(const v16h*)&wl[((tap * 2 + 0) * 2 + hh) * 256 + lr * 16];
        v16h b1 = *(const v16h*)&wl[((tap * 2 + 1) * 2 + hh) * 256 + lr * 16];
        acc0 = wmma_f16(a, b0, acc0);
        acc1 = wmma_f16(a, b1, acc1);
    }
#pragma unroll
    for (int half = 0; half < 2; ++half) {
        v8f acc = half ? acc1 : acc0;
        int ch = half * 16 + lr;
        float bv = bias[ch];
#pragma unroll
        for (int i = 0; i < 8; ++i) {
            int m = i + 8 * hh;
            size_t vox = ((size_t)(bi * DD + dc) * HH + hc) * WW + (w0 + m);
            float val = acc[i] + bv;
            if (do_lrelu) val = val > 0.f ? val : 0.01f * val;
            if (res) val += (float)res[vox * C2 + ch];
            out[vox * C2 + ch] = (_Float16)val;
            if (out2) out2[vox * C2 + ch] = (_Float16)(val + (float)res2[vox * C2 + ch]);
        }
    }
}

// ---------------- 7. final residual ----------------
__global__ __launch_bounds__(256) void final_kernel(const float* __restrict__ xnew,
                                                    const _Float16* __restrict__ y1,
                                                    const _Float16* __restrict__ y2,
                                                    float* __restrict__ out) {
    size_t idx = (size_t)blockIdx.x * 256 + threadIdx.x;  // NTOK*64
    size_t tok = idx >> 6; int c = (int)(idx & 63);
    float yv = (c < 32) ? (float)y1[tok * C2 + c] : (float)y2[tok * C2 + (c - 32)];
    out[idx] = xnew[idx] + yv;
}

// ---------------- host launch ----------------
extern "C" void kernel_launch(void* const* d_in, const int* in_sizes, int n_in,
                              void* d_out, int out_size, void* d_ws, size_t ws_size,
                              hipStream_t stream) {
    const float* x        = (const float*)d_in[0];
    const float* g1       = (const float*)d_in[2];
    const float* b1       = (const float*)d_in[3];
    const float* rpb_s    = (const float*)d_in[4];
    const float* qkv_w_s  = (const float*)d_in[5];
    const float* qkv_b_s  = (const float*)d_in[6];
    const float* proj_w_s = (const float*)d_in[7];
    const float* proj_b_s = (const float*)d_in[8];
    const float* rpb_t    = (const float*)d_in[9];
    const float* qkv_w_t  = (const float*)d_in[10];
    const float* qkv_b_t  = (const float*)d_in[11];
    const float* proj_w_t = (const float*)d_in[12];
    const float* proj_b_t = (const float*)d_in[13];
    const float* g2       = (const float*)d_in[14];
    const float* b2       = (const float*)d_in[15];
    const float* w11 = (const float*)d_in[16]; const float* bi11 = (const float*)d_in[17];
    const float* w12 = (const float*)d_in[18]; const float* bi12 = (const float*)d_in[19];
    const float* w21 = (const float*)d_in[20]; const float* bi21 = (const float*)d_in[21];
    const float* w22 = (const float*)d_in[22]; const float* bi22 = (const float*)d_in[23];

    char* ws = (char*)d_ws;
    // phase-1 buffers
    _Float16* h16     = (_Float16*)(ws + 0x0000000);   // NTOK*64*2
    _Float16* qkvs    = (_Float16*)(ws + 0x2000000);   // NTOK*192*2
    _Float16* attnout = (_Float16*)(ws + 0x7000000);   // NTOK*64*2
    float*    space   = (float*)   (ws + 0x8A00000);   // NTOK*64*4
    _Float16* qkvt    = (_Float16*)(ws + 0x2000000);   // reuse (after spatial proj)
    _Float16* ot      = (_Float16*)(ws + 0x4E00000);   // NTOK*32*2
    float*    timeb   = (float*)   (ws + 0xBE00000);   // NTOK*64*4
    float*    xnew    = (float*)   (ws + 0xF000000);   // NTOK*64*4
    // conv temps (reuse low region after combine)
    _Float16* a1  = (_Float16*)(ws + 0x0000000);
    _Float16* a2  = (_Float16*)(ws + 0x0D00000);
    _Float16* t1  = (_Float16*)(ws + 0x1A00000);
    _Float16* y1  = (_Float16*)(ws + 0x2700000);
    _Float16* a2b = (_Float16*)(ws + 0x3400000);
    _Float16* t2  = (_Float16*)(ws + 0x4100000);
    _Float16* y2  = (_Float16*)(ws + 0x4E00000);       // old ot region, dead by then

    // 1. LN1
    ln1_kernel<<<NTOK / 8, 256, 0, stream>>>(x, g1, b1, h16);
    // 2. spatial QKV GEMM
    gemm_kernel<64, 192, false><<<NTOK / 128, 256, 0, stream>>>(h16, qkv_w_s, qkv_b_s, qkvs);
    // 3. spatial window attention
    sattn_kernel<<<NWIN * NHD, 256, 0, stream>>>(qkvs, rpb_s, attnout);
    // 4. spatial projection
    gemm_kernel<64, 64, true><<<NTOK / 128, 256, 0, stream>>>(attnout, proj_w_s, proj_b_s, space);
    // 5. temporal QKV GEMM
    gemm_kernel<64, 96, false><<<NTOK / 128, 256, 0, stream>>>(h16, qkv_w_t, qkv_b_t, qkvt);
    // 6. temporal attention
    tattn_kernel<<<(NSEQ * 32) / 256, 256, 0, stream>>>(qkvt, rpb_t, ot);
    // 7. temporal projection
    gemm_kernel<32, 64, true><<<NTOK / 128, 256, 0, stream>>>(ot, proj_w_t, proj_b_t, timeb);
    // 8. combine + LN2 (writes xnew and channel-split halves)
    combine_kernel<<<NTOK / 8, 256, 0, stream>>>(x, space, timeb, g2, b2, xnew, a1, a2);
    // 9-12. residual conv blocks (implicit-GEMM WMMA)
    conv3d_kernel<<<12544 / 8, 256, 0, stream>>>(a1, w11, bi11, nullptr, t1, nullptr, nullptr, 1);
    conv3d_kernel<<<12544 / 8, 256, 0, stream>>>(t1, w12, bi12, a1, y1, a2b, a2, 0);  // y1; a2b = y1 + a2
    conv3d_kernel<<<12544 / 8, 256, 0, stream>>>(a2b, w21, bi21, nullptr, t2, nullptr, nullptr, 1);
    conv3d_kernel<<<12544 / 8, 256, 0, stream>>>(t2, w22, bi22, a2b, y2, nullptr, nullptr, 0);
    // 13. final residual
    final_kernel<<<(NTOK * 64) / 256, 256, 0, stream>>>(xnew, y1, y2, (float*)d_out);
}
